// HomogeneousGNN_89249420410962
// MI455X (gfx1250) — compile-verified, matrix operand order checked by reading, not run on hardware
//
#include <hip/hip_runtime.h>

#define DIM 512

typedef __bf16 bf16;
typedef __attribute__((ext_vector_type(16))) __bf16 v16bf;
typedef __attribute__((ext_vector_type(8)))  __bf16 v8bf;
typedef __attribute__((ext_vector_type(4)))  __bf16 v4bf;
typedef __attribute__((ext_vector_type(8)))  float  v8f;

// ---------- bf16 <-> f32 helpers (RNE) ----------
static __device__ __forceinline__ bf16 f2bf(float f) {
  unsigned u = __builtin_bit_cast(unsigned, f);
  unsigned r = u + 0x7FFFu + ((u >> 16) & 1u);
  unsigned short h = (unsigned short)(r >> 16);
  return __builtin_bit_cast(bf16, h);
}
static __device__ __forceinline__ float bf2f(bf16 b) {
  unsigned short h = __builtin_bit_cast(unsigned short, b);
  unsigned u = ((unsigned)h) << 16;
  return __builtin_bit_cast(float, u);
}

// ---------- degree count ----------
__global__ void count_kernel(const int* __restrict__ dst, float* __restrict__ cnt, int E) {
  int i = blockIdx.x * blockDim.x + threadIdx.x;
  if (i < E) atomicAdd(&cnt[dst[i]], 1.0f);
}

// ---------- edge scatter-sum (fp32 or bf16 source) ----------
template <bool BF>
__global__ void scatter_kernel(const void* __restrict__ Xv, const int* __restrict__ src,
                               const int* __restrict__ dst, float* __restrict__ summed, int E) {
  const int CH = DIM / 4;
  long long total = (long long)E * CH;
  for (long long idx = (long long)blockIdx.x * blockDim.x + threadIdx.x; idx < total;
       idx += (long long)gridDim.x * blockDim.x) {
    int e = (int)(idx / CH);
    int c = (int)(idx % CH);
    int s = src[e], d = dst[e];
    float v0, v1, v2, v3;
    if (BF) {
      const bf16* X = (const bf16*)Xv;
      v4bf q = *(const v4bf*)(X + (long long)s * DIM + c * 4);
      v0 = bf2f(q[0]); v1 = bf2f(q[1]); v2 = bf2f(q[2]); v3 = bf2f(q[3]);
    } else {
      const float* X = (const float*)Xv;
      float4 q = *((const float4*)(X + (long long)s * DIM) + c);
      v0 = q.x; v1 = q.y; v2 = q.z; v3 = q.w;
    }
    float* p = summed + (long long)d * DIM + c * 4;
    atomicAdd(p + 0, v0);
    atomicAdd(p + 1, v1);
    atomicAdd(p + 2, v2);
    atomicAdd(p + 3, v3);
  }
}

// ---------- mean (sum / max(cnt,1)) -> bf16 ----------
__global__ void mean_bf_kernel(const float* __restrict__ summed, const float* __restrict__ cnt,
                               bf16* __restrict__ out, long long total) {
  long long idx = (long long)blockIdx.x * blockDim.x + threadIdx.x;
  if (idx < total) {
    float c = fmaxf(cnt[idx >> 9], 1.0f);  // DIM == 512
    out[idx] = f2bf(summed[idx] / c);
  }
}

// ---------- fp32 -> bf16 copy ----------
__global__ void cvt_bf_kernel(const float* __restrict__ in, bf16* __restrict__ out, long long total) {
  long long idx = (long long)blockIdx.x * blockDim.x + threadIdx.x;
  if (idx < total) out[idx] = f2bf(in[idx]);
}

// ---------- weight convert + transpose: WT[n*DIM+k] = bf16(W[k*DIM+n]) ----------
__global__ void wprep_kernel(const float* __restrict__ W, bf16* __restrict__ WT) {
  int idx = blockIdx.x * blockDim.x + threadIdx.x;  // DIM*DIM
  int k = idx / DIM, n = idx % DIM;
  WT[n * DIM + k] = f2bf(W[idx]);
}

// ---------- WMMA fragment assemble from two contiguous 16B chunks ----------
// 16-bit A/B fragment per ISA layout: this lane holds K = kl..kl+7 and kl+16..kl+23
// (kl already includes +8 for lanes 16-31).
static __device__ __forceinline__ v16bf load_frag(const bf16* __restrict__ p) {
  v8bf lo = *(const v8bf*)p;
  v8bf hi = *(const v8bf*)(p + 16);
  return __builtin_shufflevector(lo, hi, 0, 1, 2, 3, 4, 5, 6, 7,
                                 8, 9, 10, 11, 12, 13, 14, 15);
}

// LDS column stride in bytes for staged weight slices: 32 K-values = 64B,
// padded to 80B to spread DS bank accesses.
#define WSTR 80

// ---------- fused SAGE GEMM: out = mean @ Wl + x @ Wr + b (optional ReLU) ----------
// block = 256 threads = 8 waves; wave -> 32 (M) x 64 (N) strip; block tile 256 x 64.
// Weight strips are double-buffered through LDS via async global->LDS DMA.
// K-loop fully unrolled: buffer index + stage guard are compile-time (no phi copies).
__global__ __launch_bounds__(256) void sage_gemm_kernel(
    const bf16* __restrict__ Amean, const bf16* __restrict__ Ax,
    const bf16* __restrict__ WlT, const bf16* __restrict__ WrT,
    const float* __restrict__ bias, float* __restrict__ outF, bf16* __restrict__ outB,
    int n_rows, int relu) {
  __shared__ __align__(16) char wsmem[2][2][64 * WSTR];  // [buf][mat][col*WSTR] = 20 KB

  const int lane = threadIdx.x & 31;
  const int wave = threadIdx.x >> 5;
  const int half = lane >> 4;   // 0: lanes 0-15, 1: lanes 16-31
  const int l16  = lane & 15;

  const int rbase = blockIdx.x * 256 + wave * 32;
  const int row0  = rbase + l16;        // M sub-tile 0
  const int row1  = rbase + 16 + l16;   // M sub-tile 1
  const int ar0 = (row0 < n_rows) ? row0 : (n_rows - 1);  // clamp: EXEC stays all-1
  const int ar1 = (row1 < n_rows) ? row1 : (n_rows - 1);
  const int col_base = blockIdx.y * 64;
  const int kadd = half * 8;

  // async-stage one 32-wide K slice of both weight strips into LDS buffer b.
  // thread t covers col = t/4, 16B chunk = t%4 (4 chunks x 16B = 64B = 32 bf16).
  const int scol   = threadIdx.x >> 2;
  const int schunk = threadIdx.x & 3;
  auto stage = [&](int b, int k0) {
    unsigned voff = (unsigned)((((col_base + scol) * DIM + k0) << 1) + schunk * 16);
    unsigned ldsl = (unsigned)(unsigned long long)(const void*)
                        &wsmem[b][0][scol * WSTR + schunk * 16];
    unsigned ldsr = (unsigned)(unsigned long long)(const void*)
                        &wsmem[b][1][scol * WSTR + schunk * 16];
    asm volatile("global_load_async_to_lds_b128 %0, %1, %2"
                 :: "v"(ldsl), "v"(voff), "s"(WlT) : "memory");
    asm volatile("global_load_async_to_lds_b128 %0, %1, %2"
                 :: "v"(ldsr), "v"(voff), "s"(WrT) : "memory");
  };

  v8f acc[2][4] = {{v8f{}, v8f{}, v8f{}, v8f{}}, {v8f{}, v8f{}, v8f{}, v8f{}}};

  stage(0, 0);  // prologue: fill buffer 0

#pragma unroll
  for (int ks = 0; ks < DIM / 32; ++ks) {
    const int b = ks & 1;
    // wait for this wave's async copies, then barrier so every wave's slice is visible
    asm volatile("s_wait_asynccnt 0x0" ::: "memory");
    __syncthreads();
    if (ks + 1 < DIM / 32) stage(b ^ 1, (ks + 1) * 32);  // overlap DMA with compute

    const int kl = ks * 32 + kadd;
    v16bf am0 = load_frag(Amean + (long long)ar0 * DIM + kl);
    v16bf ax0 = load_frag(Ax    + (long long)ar0 * DIM + kl);
    v16bf am1 = load_frag(Amean + (long long)ar1 * DIM + kl);
    v16bf ax1 = load_frag(Ax    + (long long)ar1 * DIM + kl);
#pragma unroll
    for (int j = 0; j < 4; ++j) {
      const int fb = (j * 16 + l16) * WSTR + kadd * 2;  // byte offset of this lane's frag
      v16bf bl = load_frag((const bf16*)&wsmem[b][0][fb]);
      v16bf br = load_frag((const bf16*)&wsmem[b][1][fb]);
      // B fragment reused across both M sub-tiles (4 WMMA per 2 LDS fragments)
      acc[0][j] = __builtin_amdgcn_wmma_f32_16x16x32_bf16(false, am0, false, bl,
                                                          (short)0, acc[0][j], false, false);
      acc[1][j] = __builtin_amdgcn_wmma_f32_16x16x32_bf16(false, am1, false, bl,
                                                          (short)0, acc[1][j], false, false);
      acc[0][j] = __builtin_amdgcn_wmma_f32_16x16x32_bf16(false, ax0, false, br,
                                                          (short)0, acc[0][j], false, false);
      acc[1][j] = __builtin_amdgcn_wmma_f32_16x16x32_bf16(false, ax1, false, br,
                                                          (short)0, acc[1][j], false, false);
    }
  }

  // C/D layout: VGPR i -> M = i + 8*half, N = l16
#pragma unroll
  for (int m = 0; m < 2; ++m) {
    const int out_row0 = rbase + m * 16 + half * 8;
#pragma unroll
    for (int j = 0; j < 4; ++j) {
      const int col = col_base + j * 16 + l16;
      const float bv = bias[col];
#pragma unroll
      for (int i = 0; i < 8; ++i) {
        int r = out_row0 + i;
        if (r < n_rows) {
          float v = acc[m][j][i] + bv;
          if (relu) v = fmaxf(v, 0.0f);
          if (outF) outF[(long long)r * DIM + col] = v;
          else      outB[(long long)r * DIM + col] = f2bf(v);
        }
      }
    }
  }
}

extern "C" void kernel_launch(void* const* d_in, const int* in_sizes, int n_in,
                              void* d_out, int out_size, void* d_ws, size_t ws_size,
                              hipStream_t stream) {
  const float* x   = (const float*)d_in[0];
  const int*   ei  = (const int*)d_in[1];
  const float* W1l = (const float*)d_in[2];
  const float* b1  = (const float*)d_in[3];
  const float* W1r = (const float*)d_in[4];
  const float* W2l = (const float*)d_in[5];
  const float* b2  = (const float*)d_in[6];
  const float* W2r = (const float*)d_in[7];

  const int n = in_sizes[0] / DIM;  // 50000
  const int E = in_sizes[1] / 2;    // 400000
  const int* src = ei;
  const int* dst = ei + E;
  const long long ND = (long long)n * DIM;

  // workspace carving
  char* ws = (char*)d_ws;
  size_t off = 0;
  auto carve = [&](size_t bytes) -> char* {
    char* p = ws + off;
    off += (bytes + 255) & ~(size_t)255;
    return p;
  };
  float* summed = (float*)carve((size_t)ND * 4);
  float* cnt    = (float*)carve((size_t)n * 4);
  bf16*  xb     = (bf16*)carve((size_t)ND * 2);
  bf16*  hb     = (bf16*)carve((size_t)ND * 2);
  bf16*  meanb  = (bf16*)carve((size_t)ND * 2);
  bf16*  W1lT   = (bf16*)carve((size_t)DIM * DIM * 2);
  bf16*  W1rT   = (bf16*)carve((size_t)DIM * DIM * 2);
  bf16*  W2lT   = (bf16*)carve((size_t)DIM * DIM * 2);
  bf16*  W2rT   = (bf16*)carve((size_t)DIM * DIM * 2);

  const int T = 256;
  const int wblocks = (DIM * DIM) / T;                       // 1024
  const long long nd_blocks = (ND + T - 1) / T;              // 100000
  const long long sc_blocks = ((long long)E * (DIM / 4) + T - 1) / T;
  dim3 ggrid((n + 255) / 256, DIM / 64);                     // (196, 8)

  // ---- prep ----
  hipMemsetAsync(summed, 0, (size_t)ND * 4, stream);
  hipMemsetAsync(cnt, 0, (size_t)n * 4, stream);
  wprep_kernel<<<wblocks, T, 0, stream>>>(W1l, W1lT);
  wprep_kernel<<<wblocks, T, 0, stream>>>(W1r, W1rT);
  wprep_kernel<<<wblocks, T, 0, stream>>>(W2l, W2lT);
  wprep_kernel<<<wblocks, T, 0, stream>>>(W2r, W2rT);
  cvt_bf_kernel<<<(int)nd_blocks, T, 0, stream>>>(x, xb, ND);
  count_kernel<<<(E + T - 1) / T, T, 0, stream>>>(dst, cnt, E);

  // ---- layer 1 ----
  scatter_kernel<false><<<(int)sc_blocks, T, 0, stream>>>((const void*)x, src, dst, summed, E);
  mean_bf_kernel<<<(int)nd_blocks, T, 0, stream>>>(summed, cnt, meanb, ND);
  sage_gemm_kernel<<<ggrid, T, 0, stream>>>(meanb, xb, W1lT, W1rT, b1,
                                            (float*)nullptr, hb, n, /*relu=*/1);

  // ---- layer 2 ----
  hipMemsetAsync(summed, 0, (size_t)ND * 4, stream);
  scatter_kernel<true><<<(int)sc_blocks, T, 0, stream>>>((const void*)hb, src, dst, summed, E);
  mean_bf_kernel<<<(int)nd_blocks, T, 0, stream>>>(summed, cnt, meanb, ND);
  sage_gemm_kernel<<<ggrid, T, 0, stream>>>(meanb, hb, W2lT, W2rT, b2,
                                            (float*)d_out, (bf16*)nullptr, n, /*relu=*/0);
}
